// DeformLayer_41068477284905
// MI455X (gfx1250) — matrix operand order repaired and checked by reading, not compiled, not run on hardware
//
#include <hip/hip_runtime.h>
#include <hip/hip_bf16.h>

typedef __attribute__((ext_vector_type(16))) __bf16 v16bf;
typedef __attribute__((ext_vector_type(8)))  float  v8f;

#define B_     4
#define CIN    256
#define COUT   256
#define H_     64
#define W_     64
#define HW     4096
#define KDE    2304     // 256*9
#define KT_S   72       // 2304/32
#define KT_U   32       // 1024/32
#define NTILE  256      // 4096/16
#define EPS_   1e-5f

// workspace byte offsets (all 256B aligned)
#define WS_OFFX   0u
#define WS_OFFY   589824u
#define WS_MASK   1179648u
#define WS_WDCN   1769472u      // 16*72 tiles * 1024B
#define WS_WUP    2949120u      // 4 * 16*32 tiles * 1024B
#define WS_Y1     5046272u      // 4*256*4096*4
#define WS_MEAN1  21823488u
#define WS_INV1   21824512u
#define WS_Y1N    21825536u     // bf16, 4*256*4096*2
#define WS_SPACK  30214144u     // 72*256 tiles * 1024B (per-batch reuse)
#define WS_BUP    49088512u     // 32*256 tiles * 1024B (per-batch/parity reuse)
#define WS_MEAN2  57477120u
#define WS_INV2   57478144u

// ---- CDNA5 async global->LDS staging (guarded; clean fallback if absent) ----
#if __has_builtin(__builtin_amdgcn_global_load_async_to_lds_b128)
#define HAVE_ASYNC_LDS 1
#else
#define HAVE_ASYNC_LDS 0
#endif

#if __has_builtin(__builtin_amdgcn_s_wait_asynccnt)
#define WAIT_ASYNC0() __builtin_amdgcn_s_wait_asynccnt(0)
#else
#define WAIT_ASYNC0() asm volatile("s_wait_asynccnt 0" ::: "memory")
#endif

#if HAVE_ASYNC_LDS
// builtin signature: (int4 addrspace(1)*, int4 addrspace(3)*, imm offset, cpol)
typedef int v4i_vs __attribute__((vector_size(16)));
typedef __attribute__((address_space(1))) v4i_vs gv4i_t;
typedef __attribute__((address_space(3))) v4i_vs lv4i_t;
// copy 32B per lane (one full 1KB tile per wave) global -> LDS, async
__device__ __forceinline__ void async_cp32(const char* g, unsigned ldsAddr) {
    gv4i_t* gp = (gv4i_t*)(uintptr_t)g;
    lv4i_t* lp = (lv4i_t*)(uintptr_t)ldsAddr;
    __builtin_amdgcn_global_load_async_to_lds_b128(gp, lp, 0, 0);
    __builtin_amdgcn_global_load_async_to_lds_b128(gp, lp, 16, 0);
}
#endif

__device__ __forceinline__ unsigned short f2bf(float f) {
    unsigned int u = __float_as_uint(f);
    unsigned int r = u + 0x7FFFu + ((u >> 16) & 1u);
    return (unsigned short)(r >> 16);
}

// K index within a 32-deep k-tile for A-layout (contiguous: lanes0-15 K=0..15, lanes16-31 K=16..31)
__device__ __forceinline__ int a_k_of(int d, int half, int kk) {
    return 2 * d + (half << 4) + kk;
}
// K index for B-layout (16-bit B 32x16; lanes0-15 K=0..15, lanes16-31 K=16..31)
__device__ __forceinline__ int b_k_of(int d, int half, int kk) {
    return 2 * d + (half << 4) + kk;
}

// ---------------- stage 1: offset conv (256 -> 27 ch, 3x3 pad 1) ----------------
__global__ void offset_conv_kernel(const float* __restrict__ x,
                                   const float* __restrict__ w_off,
                                   const float* __restrict__ b_off,
                                   float* __restrict__ offx,
                                   float* __restrict__ offy,
                                   float* __restrict__ maskb) {
    int gid = blockIdx.x * 256 + threadIdx.x;        // 4*27*4096 threads
    int pix = gid & (HW - 1);
    int t   = gid >> 12;
    int oc  = t % 27;
    int b   = t / 27;
    int h = pix >> 6, w = pix & 63;
    const float* xb = x + ((size_t)b * CIN << 12);
    float acc = b_off[oc];
    for (int ci = 0; ci < CIN; ++ci) {
        const float* xc = xb + (ci << 12);
        const float* wr = w_off + (oc * CIN + ci) * 9;
        #pragma unroll
        for (int kh = 0; kh < 3; ++kh) {
            int hh = h + kh - 1;
            if (hh < 0 || hh >= H_) continue;
            #pragma unroll
            for (int kw = 0; kw < 3; ++kw) {
                int ww = w + kw - 1;
                if (ww < 0 || ww >= W_) continue;
                acc = fmaf(xc[(hh << 6) + ww], wr[kh * 3 + kw], acc);
            }
        }
    }
    int oidx = ((b * 9) << 12) + pix;
    if (oc < 9)       offx[oidx + (oc << 12)] = acc;
    else if (oc < 18) offy[oidx + ((oc - 9) << 12)] = acc;
    else              maskb[oidx + ((oc - 18) << 12)] = 1.f / (1.f + expf(-acc));
}

// ---------------- pack W_dcn (256 x 2304) into bf16 WMMA A-layout ----------------
__global__ void pack_wdcn_kernel(const float* __restrict__ w_dcn, char* __restrict__ Apack) {
    int gid  = blockIdx.x * 256 + threadIdx.x;       // 16*72*32 threads
    int lane = gid & 31;
    int tile = gid >> 5;
    int mt = tile / KT_S, kt = tile - mt * KT_S;
    int m = (mt << 4) + (lane & 15);
    int half = lane >> 4;
    unsigned int u[8];
    #pragma unroll
    for (int d = 0; d < 8; ++d) {
        unsigned short lo = f2bf(w_dcn[m * KDE + (kt << 5) + a_k_of(d, half, 0)]);
        unsigned short hi = f2bf(w_dcn[m * KDE + (kt << 5) + a_k_of(d, half, 1)]);
        u[d] = (unsigned)lo | ((unsigned)hi << 16);
    }
    uint4* q = (uint4*)(Apack + ((tile << 10) + (lane << 5)));
    uint4 a; a.x = u[0]; a.y = u[1]; a.z = u[2]; a.w = u[3]; q[0] = a;
    uint4 bv; bv.x = u[4]; bv.y = u[5]; bv.z = u[6]; bv.w = u[7]; q[1] = bv;
}

// ---- pack W_up per parity class p=(ph,pw): A-layout M=256, K=1024 (ci*4 + tap) ----
__global__ void pack_wup_kernel(const float* __restrict__ w_up, char* __restrict__ Apack) {
    int gid  = blockIdx.x * 256 + threadIdx.x;       // 4*16*32*32 threads
    int lane = gid & 31;
    int tile = gid >> 5;
    int p   = tile >> 9;
    int rem = tile & 511;
    int mt = rem >> 5, kt = rem & 31;
    int ph = p >> 1, pw = p & 1;
    int m = (mt << 4) + (lane & 15);
    int half = lane >> 4;
    unsigned int u[8];
    #pragma unroll
    for (int d = 0; d < 8; ++d) {
        unsigned int uu = 0;
        #pragma unroll
        for (int kk = 0; kk < 2; ++kk) {
            int kcol = (kt << 5) + a_k_of(d, half, kk);
            int ci = kcol >> 2, tp = kcol & 3;
            int a = tp >> 1, bb = tp & 1;
            int kh = ph + 2 * a, kw = pw + 2 * bb;
            float v = w_up[(((ci * COUT) + m) * 4 + (3 - kh)) * 4 + (3 - kw)];
            uu |= ((unsigned)f2bf(v)) << (16 * kk);
        }
        u[d] = uu;
    }
    char* dst = Apack + ((size_t)p * 524288u) + (((size_t)(mt * KT_U + kt)) << 10) + (lane << 5);
    uint4* q = (uint4*)dst;
    uint4 a; a.x = u[0]; a.y = u[1]; a.z = u[2]; a.w = u[3]; q[0] = a;
    uint4 bv; bv.x = u[4]; bv.y = u[5]; bv.z = u[6]; bv.w = u[7]; q[1] = bv;
}

// ---- deformable bilinear sampling, directly into bf16 WMMA B-layout slab ----
__global__ void sample_kernel(const float* __restrict__ x,
                              const float* __restrict__ offx,
                              const float* __restrict__ offy,
                              const float* __restrict__ maskb,
                              char* __restrict__ Bpack, int b) {
    int gid  = blockIdx.x * 256 + threadIdx.x;       // 72*256*32 threads
    int lane = gid & 31;
    int tile = gid >> 5;
    int kt = tile >> 8, nt = tile & 255;
    int half = lane >> 4;
    int ncol = (nt << 4) + (lane & 15);
    int h = ncol >> 6, w = ncol & 63;
    const float* ox = offx + ((size_t)b * 9 << 12);
    const float* oy = offy + ((size_t)b * 9 << 12);
    const float* mk = maskb + ((size_t)b * 9 << 12);
    unsigned int u[8];
    #pragma unroll
    for (int d = 0; d < 8; ++d) {
        unsigned int uu = 0;
        #pragma unroll
        for (int kk = 0; kk < 2; ++kk) {
            int kRow = (kt << 5) + b_k_of(d, half, kk);
            int c = kRow / 9;
            int tap = kRow - 9 * c;
            int kh = tap / 3 - 1;
            int kw = tap - (tap / 3) * 3 - 1;
            int oidx = (tap << 12) + (h << 6) + w;
            float py = oy[oidx] + (float)(h + kh);
            float px = ox[oidx] + (float)(w + kw);
            float m1 = mk[oidx];
            float fy = floorf(py), fx = floorf(px);
            float wy = py - fy, wx = px - fx;
            int y0 = (int)fy, x0 = (int)fx;
            const float* xc = x + (((size_t)(b * CIN + c)) << 12);
            float v00 = (y0 >= 0 && y0 < H_ && x0 >= 0 && x0 < W_)         ? xc[(y0 << 6) + x0]       : 0.f;
            float v01 = (y0 >= 0 && y0 < H_ && x0+1 >= 0 && x0+1 < W_)     ? xc[(y0 << 6) + x0 + 1]   : 0.f;
            float v10 = (y0+1 >= 0 && y0+1 < H_ && x0 >= 0 && x0 < W_)     ? xc[((y0+1) << 6) + x0]   : 0.f;
            float v11 = (y0+1 >= 0 && y0+1 < H_ && x0+1 >= 0 && x0+1 < W_) ? xc[((y0+1) << 6) + x0+1] : 0.f;
            float val = (v00 * (1.f - wy) * (1.f - wx) + v01 * (1.f - wy) * wx +
                         v10 * wy * (1.f - wx)         + v11 * wy * wx) * m1;
            uu |= ((unsigned)f2bf(val)) << (16 * kk);
        }
        u[d] = uu;
    }
    uint4* q = (uint4*)(Bpack + (((size_t)(kt * NTILE + nt)) << 10) + (lane << 5));
    uint4 a; a.x = u[0]; a.y = u[1]; a.z = u[2]; a.w = u[3]; q[0] = a;
    uint4 bv; bv.x = u[4]; bv.y = u[5]; bv.z = u[6]; bv.w = u[7]; q[1] = bv;
}

// =======================================================================
// GEMM core: 8 waves, block tile 128(M) x 128(N), wave tile 64x32.
// A tiles staged block-wide through LDS via async DMA when available.
// KT = k-tile count; A tile row stride = KT*1024 bytes.
// =======================================================================
template <int KT>
__device__ __forceinline__ void gemm_core(const char* __restrict__ Apack,
                                          const char* __restrict__ Bpack,
                                          int mtBlk, int nt0, int wave, int lane,
                                          v8f acc[4][2]) {
    const int laneOff = lane << 5;
    const int mWave = wave & 1;
    const char* bcur = Bpack + ((size_t)nt0 << 10) + laneOff;   // advances 256KB/step

#if HAVE_ASYNC_LDS
    __shared__ char lds_a[2][8192];                 // 8 tiles x 1KB, double buffered
    // this wave stages block A-tile #wave each k-step
    const char* acur = Apack + (((size_t)(mtBlk + wave) * KT) << 10) + laneOff;
    const unsigned ldsW = (unsigned)(uintptr_t)(&lds_a[0][0]) + (wave << 10) + laneOff;
    async_cp32(acur, ldsW);                         // prologue: stage kt=0 -> buf 0
    const int rdOff = ((mWave * 4) << 10) + laneOff;
    for (int kt = 0; kt < KT; ++kt) {
        WAIT_ASYNC0();                              // my stage(kt) landed
        __syncthreads();                            // everyone's stage(kt) landed, prev reads done
        if (kt + 1 < KT)
            async_cp32(acur + (((size_t)(kt + 1)) << 10), ldsW + (((kt + 1) & 1) << 13));
        // read via the __shared__ object so these lower to ds_load_b128 (not flat)
        const char* ab = ((const char*)lds_a) + ((kt & 1) << 13) + rdOff;
        v16bf a[4], bv[2];
        #pragma unroll
        for (int i = 0; i < 4; ++i) a[i] = *(const v16bf*)(ab + (i << 10));
        #pragma unroll
        for (int j = 0; j < 2; ++j) bv[j] = *(const v16bf*)(bcur + (j << 10));
        #pragma unroll
        for (int i = 0; i < 4; ++i)
            #pragma unroll
            for (int j = 0; j < 2; ++j)
                acc[i][j] = __builtin_amdgcn_wmma_f32_16x16x32_bf16(
                    false, a[i], false, bv[j], (short)0, acc[i][j], false, false);
        bcur += (size_t)NTILE << 10;
    }
#else
    const char* acur = Apack + (((size_t)(mtBlk + mWave * 4) * KT) << 10) + laneOff;
    #pragma unroll 2
    for (int kt = 0; kt < KT; ++kt) {
        v16bf a[4], bv[2];
        #pragma unroll
        for (int i = 0; i < 4; ++i)
            a[i] = *(const v16bf*)(acur + (size_t)i * ((size_t)KT << 10));
        #pragma unroll
        for (int j = 0; j < 2; ++j) bv[j] = *(const v16bf*)(bcur + (j << 10));
        #pragma unroll
        for (int i = 0; i < 4; ++i)
            #pragma unroll
            for (int j = 0; j < 2; ++j)
                acc[i][j] = __builtin_amdgcn_wmma_f32_16x16x32_bf16(
                    false, a[i], false, bv[j], (short)0, acc[i][j], false, false);
        acur += 1024;
        bcur += (size_t)NTILE << 10;
    }
#endif
}

// ---------------- deform GEMM: D[256,4096] = W[256,2304] x S[2304,4096] ----------------
__global__ void __launch_bounds__(256) gemm_deform_kernel(const char* __restrict__ Apack,
                                                          const char* __restrict__ Bpack,
                                                          float* __restrict__ Yout) {
    const int lane = threadIdx.x & 31;
    const int wave = threadIdx.x >> 5;
    const int mtBlk = blockIdx.x * 8;
    const int nt0 = blockIdx.y * 8 + (wave >> 1) * 2;
    v8f acc[4][2];
    #pragma unroll
    for (int i = 0; i < 4; ++i)
        #pragma unroll
        for (int j = 0; j < 2; ++j) acc[i][j] = (v8f){};
    gemm_core<KT_S>(Apack, Bpack, mtBlk, nt0, wave, lane, acc);
    const int half = lane >> 4;
    const int col0 = (nt0 << 4) + (lane & 15);
    const int mt0 = mtBlk + (wave & 1) * 4;
    #pragma unroll
    for (int i = 0; i < 4; ++i) {
        int mb = ((mt0 + i) << 4) + (half << 3);
        #pragma unroll
        for (int j = 0; j < 2; ++j) {
            int n = col0 + (j << 4);
            #pragma unroll
            for (int r = 0; r < 8; ++r)
                Yout[((mb + r) << 12) + n] = acc[i][j][r];
        }
    }
}

// ---------------- up GEMM: D[256,4096] per parity -> scatter into (128,128) ----------------
__global__ void __launch_bounds__(256) gemm_up_kernel(const char* __restrict__ Apack,
                                                      const char* __restrict__ Bpack,
                                                      float* __restrict__ out,
                                                      int b, int ph, int pw) {
    const int lane = threadIdx.x & 31;
    const int wave = threadIdx.x >> 5;
    const int mtBlk = blockIdx.x * 8;
    const int nt0 = blockIdx.y * 8 + (wave >> 1) * 2;
    v8f acc[4][2];
    #pragma unroll
    for (int i = 0; i < 4; ++i)
        #pragma unroll
        for (int j = 0; j < 2; ++j) acc[i][j] = (v8f){};
    gemm_core<KT_U>(Apack, Bpack, mtBlk, nt0, wave, lane, acc);
    const int half = lane >> 4;
    const int col0 = (nt0 << 4) + (lane & 15);
    const int mt0 = mtBlk + (wave & 1) * 4;
    #pragma unroll
    for (int i = 0; i < 4; ++i) {
        int mb = ((mt0 + i) << 4) + (half << 3);
        #pragma unroll
        for (int j = 0; j < 2; ++j) {
            int n = col0 + (j << 4);
            int ii = n >> 6, jj = n & 63;
            int ho = 2 * ii + ph, wo = 2 * jj + pw;
            #pragma unroll
            for (int r = 0; r < 8; ++r)
                out[((size_t)(b * COUT + mb + r) << 14) + (ho << 7) + wo] = acc[i][j][r];
        }
    }
}

// ---------------- BN1 stats / apply(+relu+bf16) ----------------
__global__ void bn1_stats_kernel(const float* __restrict__ y1, float* mean, float* inv) {
    __shared__ float ss[256], sq[256];
    int c = blockIdx.x, tid = threadIdx.x;
    float s = 0.f, s2 = 0.f;
    for (int t = tid; t < B_ * HW; t += 256) {
        int b = t >> 12, n = t & (HW - 1);
        float v = y1[((size_t)(b * COUT + c) << 12) + n];
        s += v; s2 += v * v;
    }
    ss[tid] = s; sq[tid] = s2; __syncthreads();
    for (int o = 128; o > 0; o >>= 1) {
        if (tid < o) { ss[tid] += ss[tid + o]; sq[tid] += sq[tid + o]; }
        __syncthreads();
    }
    if (tid == 0) {
        float m = ss[0] / (float)(B_ * HW);
        float var = sq[0] / (float)(B_ * HW) - m * m;
        mean[c] = m; inv[c] = rsqrtf(var + EPS_);
    }
}

__global__ void bn1_apply_kernel(const float* __restrict__ y1,
                                 const float* __restrict__ mean, const float* __restrict__ inv,
                                 const float* __restrict__ gamma, const float* __restrict__ beta,
                                 unsigned short* __restrict__ y1n) {
    int idx = blockIdx.x * 256 + threadIdx.x;        // 4*256*4096
    int c = (idx >> 12) & 255;
    float v = y1[idx];
    float r = fmaxf((v - mean[c]) * inv[c] * gamma[c] + beta[c], 0.f);
    y1n[idx] = f2bf(r);
}

// ---- im2col for transposed conv, parity class (ph,pw), B-layout K=1024 ----
__global__ void im2col_up_kernel(const unsigned short* __restrict__ y1n,
                                 char* __restrict__ Bpack, int b, int ph, int pw) {
    int gid  = blockIdx.x * 256 + threadIdx.x;       // 32*256*32 threads
    int lane = gid & 31;
    int tile = gid >> 5;
    int kt = tile >> 8, nt = tile & 255;
    int half = lane >> 4;
    int ncol = (nt << 4) + (lane & 15);
    int ii = ncol >> 6, jj = ncol & 63;
    unsigned int u[8];
    #pragma unroll
    for (int d = 0; d < 8; ++d) {
        unsigned int uu = 0;
        #pragma unroll
        for (int kk = 0; kk < 2; ++kk) {
            int kRow = (kt << 5) + b_k_of(d, half, kk);
            int ci = kRow >> 2, tp = kRow & 3;
            int a = tp >> 1, bb = tp & 1;
            int i = ii + ph + a - 1;
            int j = jj + pw + bb - 1;
            unsigned short v = 0;
            if (i >= 0 && i < H_ && j >= 0 && j < W_)
                v = y1n[((size_t)(b * COUT + ci) << 12) + (i << 6) + j];
            uu |= ((unsigned)v) << (16 * kk);
        }
        u[d] = uu;
    }
    uint4* q = (uint4*)(Bpack + (((size_t)(kt * NTILE + nt)) << 10) + (lane << 5));
    uint4 a; a.x = u[0]; a.y = u[1]; a.z = u[2]; a.w = u[3]; q[0] = a;
    uint4 bv; bv.x = u[4]; bv.y = u[5]; bv.z = u[6]; bv.w = u[7]; q[1] = bv;
}

// ---------------- BN2 stats / apply(+relu) in place on d_out ----------------
__global__ void bn2_stats_kernel(const float* __restrict__ y, float* mean, float* inv) {
    __shared__ float ss[256], sq[256];
    int c = blockIdx.x, tid = threadIdx.x;
    float s = 0.f, s2 = 0.f;
    for (int t = tid; t < B_ * 16384; t += 256) {
        int b = t >> 14, n = t & 16383;
        float v = y[((size_t)(b * COUT + c) << 14) + n];
        s += v; s2 += v * v;
    }
    ss[tid] = s; sq[tid] = s2; __syncthreads();
    for (int o = 128; o > 0; o >>= 1) {
        if (tid < o) { ss[tid] += ss[tid + o]; sq[tid] += sq[tid + o]; }
        __syncthreads();
    }
    if (tid == 0) {
        float m = ss[0] / (float)(B_ * 16384);
        float var = sq[0] / (float)(B_ * 16384) - m * m;
        mean[c] = m; inv[c] = rsqrtf(var + EPS_);
    }
}

__global__ void bn2_apply_kernel(float* __restrict__ y,
                                 const float* __restrict__ mean, const float* __restrict__ inv,
                                 const float* __restrict__ gamma, const float* __restrict__ beta) {
    size_t idx = (size_t)blockIdx.x * 256 + threadIdx.x;   // 4*256*16384
    int c = (int)((idx >> 14) & 255);
    float v = y[idx];
    y[idx] = fmaxf((v - mean[c]) * inv[c] * gamma[c] + beta[c], 0.f);
}

extern "C" void kernel_launch(void* const* d_in, const int* in_sizes, int n_in,
                              void* d_out, int out_size, void* d_ws, size_t ws_size,
                              hipStream_t stream) {
    (void)in_sizes; (void)n_in; (void)out_size; (void)ws_size;
    const float* x     = (const float*)d_in[0];
    const float* w_off = (const float*)d_in[1];
    const float* b_off = (const float*)d_in[2];
    const float* w_dcn = (const float*)d_in[3];
    const float* g1    = (const float*)d_in[4];
    const float* be1   = (const float*)d_in[5];
    const float* w_up  = (const float*)d_in[6];
    const float* g2    = (const float*)d_in[7];
    const float* be2   = (const float*)d_in[8];
    char*  ws  = (char*)d_ws;
    float* out = (float*)d_out;

    float* offx = (float*)(ws + WS_OFFX);
    float* offy = (float*)(ws + WS_OFFY);
    float* mskb = (float*)(ws + WS_MASK);
    char*  wdcnP = ws + WS_WDCN;
    char*  wupP  = ws + WS_WUP;
    float* y1    = (float*)(ws + WS_Y1);
    float* mean1 = (float*)(ws + WS_MEAN1);
    float* inv1  = (float*)(ws + WS_INV1);
    unsigned short* y1n = (unsigned short*)(ws + WS_Y1N);
    char*  spack = ws + WS_SPACK;
    char*  bupP  = ws + WS_BUP;
    float* mean2 = (float*)(ws + WS_MEAN2);
    float* inv2  = (float*)(ws + WS_INV2);

    offset_conv_kernel<<<1728, 256, 0, stream>>>(x, w_off, b_off, offx, offy, mskb);
    pack_wdcn_kernel<<<144, 256, 0, stream>>>(w_dcn, wdcnP);
    pack_wup_kernel<<<256, 256, 0, stream>>>(w_up, wupP);

    for (int b = 0; b < B_; ++b) {
        sample_kernel<<<2304, 256, 0, stream>>>(x, offx, offy, mskb, spack, b);
        gemm_deform_kernel<<<dim3(2, 32), 256, 0, stream>>>(wdcnP, spack,
                                                            y1 + ((size_t)b * COUT << 12));
    }

    bn1_stats_kernel<<<256, 256, 0, stream>>>(y1, mean1, inv1);
    bn1_apply_kernel<<<16384, 256, 0, stream>>>(y1, mean1, inv1, g1, be1, y1n);

    for (int b = 0; b < B_; ++b) {
        for (int p = 0; p < 4; ++p) {
            int ph = p >> 1, pw = p & 1;
            im2col_up_kernel<<<1024, 256, 0, stream>>>(y1n, bupP, b, ph, pw);
            gemm_up_kernel<<<dim3(2, 32), 256, 0, stream>>>(wupP + (size_t)p * 524288u,
                                                            bupP, out, b, ph, pw);
        }
    }

    bn2_stats_kernel<<<256, 256, 0, stream>>>(out, mean2, inv2);
    bn2_apply_kernel<<<65536, 256, 0, stream>>>(out, mean2, inv2, g2, be2);
}
